// LabelComp_75600014344330
// MI455X (gfx1250) — compile-verified, matrix operand order checked by reading, not run on hardware
//
#include <hip/hip_runtime.h>
#include <hip/hip_bf16.h>
#include <stdint.h>

// Reference constants
#define HOPSZ   160
#define FFTW    1024
#define NLAB    72
#define NBATCH  4
#define TLEN    240000
#define PADW    512            // FFT/2
#define NOUT    1501           // (T + 2*pad - FFT)/HOP + 1
#define WAVES_PER_BLOCK 8

typedef int v8i __attribute__((ext_vector_type(8)));

__global__ __launch_bounds__(256)
void LabelComp_majority_kernel(const int* __restrict__ lbl, float* __restrict__ out) {
    // 8 waves/block, 1 KB of packed label bytes per wave (256 dwords)
    __shared__ uint32_t slab[WAVES_PER_BLOCK * (FFTW / 4)];

    const int lane  = threadIdx.x & 31;
    const int wave  = threadIdx.x >> 5;
    const int gwave = blockIdx.x * WAVES_PER_BLOCK + wave;
    const bool active = (gwave < NBATCH * NOUT);
    const int gw = active ? gwave : 0;          // inactive waves do dummy work on window 0
    const int b  = gw / NOUT;
    const int w  = gw - b * NOUT;
    const int p0 = w * HOPSZ;                   // window start in padded coordinates
    const int srcbase = b * TLEN;

    uint32_t* myl = &slab[wave * (FFTW / 4)];

    // ---- Stage 1024 reflect-padded labels into LDS as packed bytes ----
    #pragma unroll
    for (int i = 0; i < 8; ++i) {
        const int d  = i * 32 + lane;           // dword index 0..255
        const int e0 = d * 4;
        uint32_t pack = 0;
        #pragma unroll
        for (int j = 0; j < 4; ++j) {
            int src = p0 + e0 + j - PADW;       // reflect (no edge repeat), pad < T
            src = (src < 0) ? -src : src;
            src = (src >= TLEN) ? (2 * (TLEN - 1) - src) : src;
            pack |= (uint32_t)(lbl[srcbase + src] & 0xFF) << (8 * j);
        }
        myl[d] = pack;
    }
    __syncthreads();

    // ---- WMMA count accumulation: D = A(indicator 16x64 iu8) x B(ones 64x16) + C ----
    const int half = (lane >> 4) & 1;           // which K-half of each A row this lane supplies
    const int mrow = lane & 15;                 // A row (channel within group) this lane builds

    v8i ones;
    #pragma unroll
    for (int r = 0; r < 8; ++r) ones[r] = 0x01010101;

    v8i acc[5];
    #pragma unroll
    for (int g = 0; g < 5; ++g)
        #pragma unroll
        for (int r = 0; r < 8; ++r) acc[g][r] = 0;

    // 8-bit A 16x64 VGPR layout (ISA 7.12.2): lanes 0-15 K-chunks {0-7,16-23,32-39,48-55},
    // lanes 16-31 K-chunks {8-15,24-31,40-47,56-63}; dword-of-K offsets within a segment:
    const int kd[8] = {0, 1, 4, 5, 8, 9, 12, 13};   // dword units; +2 dwords for upper half

    for (int s = 0; s < 16; ++s) {
        uint32_t labs[8];
        const int segbase = s * 16 + half * 2;      // dword index into myl
        #pragma unroll
        for (int j = 0; j < 8; ++j) labs[j] = myl[segbase + kd[j]];

        // Build ALL five indicator matrices into distinct register buffers first,
        // then issue the 5 WMMAs back-to-back. No WAR on A operands within a
        // segment -> no IU8 WMMA->VALU hazard NOPs, and XDL WMMAs can co-execute
        // with the next segment's VALU compare work.
        v8i Abuf[5];
        #pragma unroll
        for (int g = 0; g < 5; ++g) {
            const uint32_t cpat = (uint32_t)(16 * g + mrow) * 0x01010101u;
            #pragma unroll
            for (int j = 0; j < 8; ++j) {
                // bytes of (labs ^ cpat) are in [0,0x7F]; +0x7F per byte never carries,
                // MSB stays 0 only where the byte was 0 (i.e. label == channel)
                const uint32_t x = labs[j] ^ cpat;
                const uint32_t m = ((~(x + 0x7F7F7F7Fu)) & 0x80808080u) >> 7;
                Abuf[g][j] = (int)m;                 // 0x01 at matches
            }
        }
        #pragma unroll
        for (int g = 0; g < 5; ++g) {
            acc[g] = __builtin_amdgcn_wmma_i32_16x16x64_iu8(
                false, Abuf[g], false, ones, acc[g], false, false);
        }
    }

    // ---- Extract counts and argmax (lowest label wins ties) ----
    // 32-bit D layout: lanes 0-15 hold rows 0-7 (VGPR r -> row r), lanes 16-31 rows 8-15.
    int bestkey = 0;
    #pragma unroll
    for (int g = 0; g < 5; ++g) {
        #pragma unroll
        for (int r = 0; r < 8; ++r) {
            const int c   = 16 * g + half * 8 + r;
            const int cnt = acc[g][r];
            const int key = (c < NLAB) ? ((cnt << 7) | (127 - c)) : 0;
            bestkey = (key > bestkey) ? key : bestkey;
        }
    }
    #pragma unroll
    for (int m = 16; m >= 1; m >>= 1) {
        const int o = __shfl_xor(bestkey, m, 32);
        bestkey = (o > bestkey) ? o : bestkey;
    }

    if (active && lane == 0)
        out[gw] = (float)(127 - (bestkey & 127));
}

extern "C" void kernel_launch(void* const* d_in, const int* in_sizes, int n_in,
                              void* d_out, int out_size, void* d_ws, size_t ws_size,
                              hipStream_t stream) {
    (void)in_sizes; (void)n_in; (void)out_size; (void)d_ws; (void)ws_size;
    const int* lbl = (const int*)d_in[0];     // [4, 240000] int32 labels
    // d_in[1] = all-ones fp32 conv weight (frozen) -> counting, not needed
    float* out = (float*)d_out;               // [4, 1501] argmax labels as float

    const int total  = NBATCH * NOUT;                                   // 6004 windows
    const int blocks = (total + WAVES_PER_BLOCK - 1) / WAVES_PER_BLOCK; // 751
    hipLaunchKernelGGL(LabelComp_majority_kernel, dim3(blocks), dim3(256), 0, stream,
                       lbl, out);
}
